// LSTMGCN_48473000903083
// MI455X (gfx1250) — compile-verified, matrix operand order checked by reading, not run on hardware
//
#include <hip/hip_runtime.h>
#include <hip/hip_bf16.h>
#include <math.h>

// ---------------------------------------------------------------------------
// Types / constants
// ---------------------------------------------------------------------------
typedef __bf16 bf16_t;
typedef __attribute__((ext_vector_type(16))) __bf16 v16bf;
typedef __attribute__((ext_vector_type(8)))  float  v8f;

#define N_SEQ  2048
#define T_LEN  256
#define D_IN   16
#define H_DIM  256
#define G4     1024
#define HG_DIM 256
#define OG_DIM 128

static __device__ __forceinline__ v8f wmma_bf16(v16bf a, v16bf b, v8f c) {
  // D = A(16x32 bf16) * B(32x16 bf16) + C(16x16 f32)
  return __builtin_amdgcn_wmma_f32_16x16x32_bf16(
      /*neg_a=*/false, a, /*neg_b=*/false, b,
      /*c_mod=*/(short)0, c, /*reuse_a=*/false, /*reuse_b=*/false);
}

union FragU { v16bf v; uint4 q[2]; bf16_t e[16]; };

static __device__ __forceinline__ v8f splat8(float x) {
  v8f r;
#pragma unroll
  for (int i = 0; i < 8; ++i) r[i] = x;
  return r;
}

// Load a 16x32 bf16 fragment (A-style: lane = row, K striped) or a 32x16
// B-fragment (lane = column of B == row of Bt, K striped) from row-major
// bf16 memory with row stride `rs`. koff = 0 for lanes 0-15, 8 for 16-31.
static __device__ __forceinline__ v16bf load_frag(const bf16_t* __restrict__ base,
                                                  int rs, int row, int k0, int koff) {
  FragU u;
  const bf16_t* p = base + (size_t)row * rs + k0 + koff;
  u.q[0] = *(const uint4*)(p);        // K = k0+koff .. +7
  u.q[1] = *(const uint4*)(p + 16);   // K = k0+16+koff .. +7
  return u.v;
}

static __device__ __forceinline__ float sigf(float x) {
  return 1.0f / (1.0f + __expf(-x));
}

static __device__ __forceinline__ float fast_tanh(float x) {
  x = fminf(15.0f, fmaxf(-15.0f, x));
  float e = __expf(2.0f * x);
  return (e - 1.0f) / (e + 1.0f);
}

// ---------------------------------------------------------------------------
// Fused 2-layer LSTM. One workgroup = 16 sequences, 512 threads = 16 waves,
// full T loop. Wave w owns hidden columns [16w, 16w+16) and all 4 gates for
// them (one 16x16 WMMA tile per gate) -> cell update is fully in-register.
// Weights are streamed from L2 every timestep; the laundered zero offset
// defeats LICM (no hoist->spill) while keeping global-addrspace inference
// (global_load_b128, not flat_load_b128). Each k-step batches all four
// B-fragment loads before the four WMMAs so loads overlap WMMA issue.
// ---------------------------------------------------------------------------
__global__ __launch_bounds__(512, 2) void k_lstm_fused(
    const float* __restrict__ x,                                   // [N,T,D]
    const bf16_t* __restrict__ wih0, const bf16_t* __restrict__ whh0,
    const bf16_t* __restrict__ wih1, const bf16_t* __restrict__ whh1,
    const float* __restrict__ b0, const float* __restrict__ b1,    // [1024]
    float* __restrict__ nf, bf16_t* __restrict__ nf_b)             // [N,H]
{
  __shared__ bf16_t h0s[16 * H_DIM];
  __shared__ bf16_t h1s[16 * H_DIM];

  const int tid   = threadIdx.x;
  const int wave  = tid >> 5;              // 0..15
  const int lane  = tid & 31;
  const int ln    = lane & 15;
  const int koff  = (lane < 16) ? 0 : 8;
  const int mrow  = (lane >> 4) * 8;       // C-layout row base for this lane
  const int bbase = blockIdx.x * 16;
  const int col   = wave * 16 + ln;        // hidden column owned by this lane

  for (int i = tid; i < 16 * H_DIM; i += 512) {
    h0s[i] = (bf16_t)0.0f;
    h1s[i] = (bf16_t)0.0f;
  }

  // Gate rows / biases owned by this lane: one tile per gate g.
  int   rowb[4];
  float bias0[4], bias1[4];
#pragma unroll
  for (int g = 0; g < 4; ++g) {
    int r = g * 256 + col;
    rowb[g]  = r;
    bias0[g] = b0[r];
    bias1[g] = b1[r];
  }

  // Preload Wih0 B-fragments once (K padded 16->32 with zeros): 32 VGPRs.
  v16bf w0frag[4];
#pragma unroll
  for (int g = 0; g < 4; ++g) {
    FragU u;
    const bf16_t* p = wih0 + (size_t)rowb[g] * D_IN + koff;
    u.q[0] = *(const uint4*)p;
    uint4 z; z.x = z.y = z.z = z.w = 0u;
    u.q[1] = z;                           // K 16..31 = 0
    w0frag[g] = u.v;
  }

  v8f c0 = splat8(0.0f), c1 = splat8(0.0f);

  __syncthreads();

  for (int t = 0; t < T_LEN; ++t) {
    // Opaque zero offset: addresses below are not provably loop-invariant,
    // so the ~96 fragment loads stay inside the loop (global_load_b128 from
    // L2) instead of being hoisted and spilled. GEP on the kernarg pointer
    // preserves global addrspace inference.
    size_t woff = 0;
    asm volatile("" : "+s"(woff));
    const bf16_t* whh0p = whh0 + woff;
    const bf16_t* wih1p = wih1 + woff;
    const bf16_t* whh1p = whh1 + woff;

    // ----------------------------- layer 0 ------------------------------
    v8f acc[4];
#pragma unroll
    for (int g = 0; g < 4; ++g) acc[g] = splat8(bias0[g]);

    // x_t @ Wih0^T : one zero-padded K=32 WMMA per gate tile
    {
      FragU ua;
      const float* xp = x + ((size_t)(bbase + ln) * T_LEN + t) * D_IN + koff;
      float4 f0 = *(const float4*)(xp);
      float4 f1 = *(const float4*)(xp + 4);
      ua.e[0] = (bf16_t)f0.x; ua.e[1] = (bf16_t)f0.y;
      ua.e[2] = (bf16_t)f0.z; ua.e[3] = (bf16_t)f0.w;
      ua.e[4] = (bf16_t)f1.x; ua.e[5] = (bf16_t)f1.y;
      ua.e[6] = (bf16_t)f1.z; ua.e[7] = (bf16_t)f1.w;
      uint4 z; z.x = z.y = z.z = z.w = 0u;
      ua.q[1] = z;
#pragma unroll
      for (int g = 0; g < 4; ++g) acc[g] = wmma_bf16(ua.v, w0frag[g], acc[g]);
    }

    // h0_{t-1} @ Whh0^T : batch the 4 B-frag loads, then 4 WMMAs
#pragma unroll
    for (int ks = 0; ks < 8; ++ks) {
      v16bf af = load_frag(h0s, H_DIM, ln, ks * 32, koff);
      v16bf bm[4];
#pragma unroll
      for (int g = 0; g < 4; ++g)
        bm[g] = load_frag(whh0p, H_DIM, rowb[g], ks * 32, koff);
#pragma unroll
      for (int g = 0; g < 4; ++g) acc[g] = wmma_bf16(af, bm[g], acc[g]);
    }

    __syncthreads();   // all reads of old h0s done

    // in-register cell update (wave owns all 4 gates of its hidden slice)
#pragma unroll
    for (int r = 0; r < 8; ++r) {
      float cn = sigf(acc[1][r]) * c0[r] + sigf(acc[0][r]) * fast_tanh(acc[2][r]);
      float hn = sigf(acc[3][r]) * fast_tanh(cn);
      c0[r] = cn;
      h0s[(mrow + r) * H_DIM + col] = (bf16_t)hn;
    }
    __syncthreads();   // new h0 visible (input of layer 1)

    // ----------------------------- layer 1 ------------------------------
#pragma unroll
    for (int g = 0; g < 4; ++g) acc[g] = splat8(bias1[g]);

#pragma unroll
    for (int ks = 0; ks < 8; ++ks) {
      // h0_t @ Wih1^T : batch loads, then WMMAs
      v16bf af0 = load_frag(h0s, H_DIM, ln, ks * 32, koff);
      v16bf bi[4];
#pragma unroll
      for (int g = 0; g < 4; ++g)
        bi[g] = load_frag(wih1p, H_DIM, rowb[g], ks * 32, koff);
#pragma unroll
      for (int g = 0; g < 4; ++g) acc[g] = wmma_bf16(af0, bi[g], acc[g]);

      // h1_{t-1} @ Whh1^T : batch loads, then WMMAs
      v16bf af1 = load_frag(h1s, H_DIM, ln, ks * 32, koff);
      v16bf bh[4];
#pragma unroll
      for (int g = 0; g < 4; ++g)
        bh[g] = load_frag(whh1p, H_DIM, rowb[g], ks * 32, koff);
#pragma unroll
      for (int g = 0; g < 4; ++g) acc[g] = wmma_bf16(af1, bh[g], acc[g]);
    }

    __syncthreads();

#pragma unroll
    for (int r = 0; r < 8; ++r) {
      float cn = sigf(acc[1][r]) * c1[r] + sigf(acc[0][r]) * fast_tanh(acc[2][r]);
      float hn = sigf(acc[3][r]) * fast_tanh(cn);
      c1[r] = cn;
      h1s[(mrow + r) * H_DIM + col] = (bf16_t)hn;
      if (t == T_LEN - 1) {
        size_t gi = (size_t)(bbase + mrow + r) * H_DIM + col;
        nf[gi]   = hn;
        nf_b[gi] = (bf16_t)hn;
      }
    }
    __syncthreads();
  }
}

// ---------------------------------------------------------------------------
// Generic WMMA GEMM: C[MxN] = A[MxK] * Bt[NxK]^T, bf16 in, f32 accumulate.
// One wave computes a 16x64 tile. mode bits: 1=relu, 2=store f32 C,
// 4=store bf16 C, 8=store bf16 C transposed (feeds next GEMM's Bt).
// ---------------------------------------------------------------------------
__global__ __launch_bounds__(256) void k_gemm(
    const bf16_t* __restrict__ A, const bf16_t* __restrict__ Bt,
    int M, int N, int K,
    float* __restrict__ Cf, bf16_t* __restrict__ Cb, bf16_t* __restrict__ Cbt,
    int mode)
{
  const int tid  = threadIdx.x;
  const int wave = tid >> 5;
  const int lane = tid & 31;
  const int ln   = lane & 15;
  const int koff = (lane < 16) ? 0 : 8;
  const int mrow = (lane >> 4) * 8;

  const int W      = blockIdx.x * 8 + wave;
  const int tilesN = N >> 6;
  const int tm     = W / tilesN;
  const int tn     = W % tilesN;
  if (tm * 16 >= M) return;

  v8f acc[4];
#pragma unroll
  for (int s = 0; s < 4; ++s) acc[s] = splat8(0.0f);

  const int arow = tm * 16 + ln;
  for (int k0 = 0; k0 < K; k0 += 32) {
    v16bf af = load_frag(A, K, arow, k0, koff);
    v16bf bm[4];
#pragma unroll
    for (int s = 0; s < 4; ++s)
      bm[s] = load_frag(Bt, K, tn * 64 + s * 16 + ln, k0, koff);
#pragma unroll
    for (int s = 0; s < 4; ++s) acc[s] = wmma_bf16(af, bm[s], acc[s]);
  }

#pragma unroll
  for (int s = 0; s < 4; ++s) {
    int n = tn * 64 + s * 16 + ln;
#pragma unroll
    for (int r = 0; r < 8; ++r) {
      int m = tm * 16 + mrow + r;
      float v = acc[s][r];
      if (mode & 1) v = v > 0.0f ? v : 0.0f;
      if (mode & 2) Cf[(size_t)m * N + n] = v;
      if (mode & 4) Cb[(size_t)m * N + n] = (bf16_t)v;
      if (mode & 8) Cbt[(size_t)n * M + m] = (bf16_t)v;
    }
  }
}

// ---------------------------------------------------------------------------
// Small helper kernels
// ---------------------------------------------------------------------------
__global__ void k_cvt(const float* __restrict__ s, bf16_t* __restrict__ d, int n) {
  int i = blockIdx.x * blockDim.x + threadIdx.x;
  if (i < n) d[i] = (bf16_t)s[i];
}

__global__ void k_cvt_t(const float* __restrict__ s, bf16_t* __restrict__ d,
                        int rows, int cols) {  // d[c][r] = s[r][c]
  int i = blockIdx.x * blockDim.x + threadIdx.x;
  if (i < rows * cols) {
    int r = i / cols, c = i % cols;
    d[(size_t)c * rows + r] = (bf16_t)s[i];
  }
}

__global__ void k_bias_add(const float* __restrict__ a, const float* __restrict__ b,
                           float* __restrict__ o, int n) {
  int i = blockIdx.x * blockDim.x + threadIdx.x;
  if (i < n) o[i] = a[i] + b[i];
}

__global__ __launch_bounds__(256) void k_meansub(const float* __restrict__ nf,
                                                 bf16_t* __restrict__ xc_b) {
  __shared__ float red[256];
  int row = blockIdx.x;
  float v = nf[(size_t)row * H_DIM + threadIdx.x];
  red[threadIdx.x] = v;
  __syncthreads();
  for (int off = 128; off > 0; off >>= 1) {
    if (threadIdx.x < off) red[threadIdx.x] += red[threadIdx.x + off];
    __syncthreads();
  }
  float mean = red[0] * (1.0f / H_DIM);
  xc_b[(size_t)row * H_DIM + threadIdx.x] = (bf16_t)(v - mean);
}

__global__ void k_diag(const float* __restrict__ cov, float* __restrict__ dvec) {
  int i = blockIdx.x * blockDim.x + threadIdx.x;
  if (i < N_SEQ) dvec[i] = sqrtf(cov[(size_t)i * N_SEQ + i]);
}

static __device__ __forceinline__ float corr_at(const float* __restrict__ cov,
                                                const float* __restrict__ dvec,
                                                int i, int j) {
  float den = dvec[i] * dvec[j];
  float cr  = cov[(size_t)i * N_SEQ + j] / den;
  if (!isfinite(cr)) cr = 0.0f;           // nan_to_num
  cr = fminf(1.0f, fmaxf(-1.0f, cr));     // clip
  if (i == j) cr += 1.0f;                 // + I
  return cr;
}

__global__ __launch_bounds__(256) void k_rowsum_dinv(const float* __restrict__ cov,
                                                     const float* __restrict__ dvec,
                                                     float* __restrict__ dinv) {
  __shared__ float red[256];
  int row = blockIdx.x;
  float s = 0.0f;
  for (int j = threadIdx.x; j < N_SEQ; j += 256) s += corr_at(cov, dvec, row, j);
  red[threadIdx.x] = s;
  __syncthreads();
  for (int off = 128; off > 0; off >>= 1) {
    if (threadIdx.x < off) red[threadIdx.x] += red[threadIdx.x + off];
    __syncthreads();
  }
  if (threadIdx.x == 0) {
    float r = rsqrtf(red[0]);             // sum^-0.5
    dinv[row] = isfinite(r) ? r : 0.0f;   // isinf -> 0
  }
}

__global__ __launch_bounds__(256) void k_adj(const float* __restrict__ cov,
                                             const float* __restrict__ dvec,
                                             const float* __restrict__ dinv,
                                             bf16_t* __restrict__ adj) {
  int row  = blockIdx.x;
  float di = dinv[row];
  for (int j = threadIdx.x; j < N_SEQ; j += 256)
    adj[(size_t)row * N_SEQ + j] = (bf16_t)(di * corr_at(cov, dvec, row, j) * dinv[j]);
}

__global__ void k_fc(const float* __restrict__ g2, const float* __restrict__ wfc,
                     const float* __restrict__ bfc, float* __restrict__ out) {
  int i = blockIdx.x * blockDim.x + threadIdx.x;
  if (i < N_SEQ) {
    float s = bfc[0];
#pragma unroll 8
    for (int j = 0; j < OG_DIM; ++j) s += g2[(size_t)i * OG_DIM + j] * wfc[j];
    out[i] = s;
  }
}

// ---------------------------------------------------------------------------
// Host driver
// ---------------------------------------------------------------------------
static inline int gemm_blocks(int M, int N) { return (M / 16) * (N / 64) / 8; }

extern "C" void kernel_launch(void* const* d_in, const int* in_sizes, int n_in,
                              void* d_out, int out_size, void* d_ws, size_t ws_size,
                              hipStream_t stream) {
  (void)in_sizes; (void)n_in; (void)out_size; (void)ws_size;
  const float* x    = (const float*)d_in[0];
  const float* Wih0 = (const float*)d_in[1];
  const float* Whh0 = (const float*)d_in[2];
  const float* bih0 = (const float*)d_in[3];
  const float* bhh0 = (const float*)d_in[4];
  const float* Wih1 = (const float*)d_in[5];
  const float* Whh1 = (const float*)d_in[6];
  const float* bih1 = (const float*)d_in[7];
  const float* bhh1 = (const float*)d_in[8];
  const float* Wg1  = (const float*)d_in[9];
  const float* Wg2  = (const float*)d_in[10];
  const float* Wfc  = (const float*)d_in[11];
  const float* bfc  = (const float*)d_in[12];

  char* ws = (char*)d_ws;
  size_t o = 0;
  auto alloc = [&](size_t bytes) -> void* {
    void* p = ws + o;
    o += (bytes + 255) & ~(size_t)255;
    return p;
  };

  bf16_t* wih0b = (bf16_t*)alloc((size_t)G4 * D_IN * 2);
  bf16_t* whh0b = (bf16_t*)alloc((size_t)G4 * H_DIM * 2);
  bf16_t* wih1b = (bf16_t*)alloc((size_t)G4 * H_DIM * 2);
  bf16_t* whh1b = (bf16_t*)alloc((size_t)G4 * H_DIM * 2);
  float*  b0    = (float*) alloc((size_t)G4 * 4);
  float*  b1    = (float*) alloc((size_t)G4 * 4);
  bf16_t* wg1t  = (bf16_t*)alloc((size_t)HG_DIM * H_DIM * 2);   // [HG][H]
  bf16_t* wg2t  = (bf16_t*)alloc((size_t)OG_DIM * HG_DIM * 2);  // [OG][HG]
  float*  nf    = (float*) alloc((size_t)N_SEQ * H_DIM * 4);
  bf16_t* nfb   = (bf16_t*)alloc((size_t)N_SEQ * H_DIM * 2);
  bf16_t* xcb   = (bf16_t*)alloc((size_t)N_SEQ * H_DIM * 2);
  float*  dvec  = (float*) alloc((size_t)N_SEQ * 4);
  float*  dinv  = (float*) alloc((size_t)N_SEQ * 4);
  float*  cov   = (float*) alloc((size_t)N_SEQ * N_SEQ * 4);
  bf16_t* adjb  = (bf16_t*)alloc((size_t)N_SEQ * N_SEQ * 2);
  bf16_t* t1t   = (bf16_t*)alloc((size_t)HG_DIM * N_SEQ * 2);   // (nf@Wg1)^T
  bf16_t* g1b   = (bf16_t*)alloc((size_t)N_SEQ * HG_DIM * 2);
  bf16_t* t2t   = (bf16_t*)alloc((size_t)OG_DIM * N_SEQ * 2);   // (g1@Wg2)^T
  float*  g2    = (float*) alloc((size_t)N_SEQ * OG_DIM * 4);

  // --- weight prep ---
  k_cvt<<<(G4 * D_IN + 255) / 256, 256, 0, stream>>>(Wih0, wih0b, G4 * D_IN);
  k_cvt<<<(G4 * H_DIM + 255) / 256, 256, 0, stream>>>(Whh0, whh0b, G4 * H_DIM);
  k_cvt<<<(G4 * H_DIM + 255) / 256, 256, 0, stream>>>(Wih1, wih1b, G4 * H_DIM);
  k_cvt<<<(G4 * H_DIM + 255) / 256, 256, 0, stream>>>(Whh1, whh1b, G4 * H_DIM);
  k_bias_add<<<(G4 + 255) / 256, 256, 0, stream>>>(bih0, bhh0, b0, G4);
  k_bias_add<<<(G4 + 255) / 256, 256, 0, stream>>>(bih1, bhh1, b1, G4);
  k_cvt_t<<<(H_DIM * HG_DIM + 255) / 256, 256, 0, stream>>>(Wg1, wg1t, H_DIM, HG_DIM);
  k_cvt_t<<<(HG_DIM * OG_DIM + 255) / 256, 256, 0, stream>>>(Wg2, wg2t, HG_DIM, OG_DIM);

  // --- fused 2-layer LSTM over all timesteps ---
  k_lstm_fused<<<N_SEQ / 16, 512, 0, stream>>>(x, wih0b, whh0b, wih1b, whh1b,
                                               b0, b1, nf, nfb);

  // --- adjacency ---
  k_meansub<<<N_SEQ, 256, 0, stream>>>(nf, xcb);
  k_gemm<<<gemm_blocks(N_SEQ, N_SEQ), 256, 0, stream>>>(
      xcb, xcb, N_SEQ, N_SEQ, H_DIM, cov, nullptr, nullptr, /*mode=*/2);
  k_diag<<<(N_SEQ + 255) / 256, 256, 0, stream>>>(cov, dvec);
  k_rowsum_dinv<<<N_SEQ, 256, 0, stream>>>(cov, dvec, dinv);
  k_adj<<<N_SEQ, 256, 0, stream>>>(cov, dvec, dinv, adjb);

  // --- GCN ---
  // t1 = nf @ Wg1   (store transposed bf16 -> Bt of next GEMM)
  k_gemm<<<gemm_blocks(N_SEQ, HG_DIM), 256, 0, stream>>>(
      nfb, wg1t, N_SEQ, HG_DIM, H_DIM, nullptr, nullptr, t1t, /*mode=*/8);
  // g1 = relu(adj @ t1)   (store bf16 row-major -> A of next GEMM)
  k_gemm<<<gemm_blocks(N_SEQ, HG_DIM), 256, 0, stream>>>(
      adjb, t1t, N_SEQ, HG_DIM, N_SEQ, nullptr, g1b, nullptr, /*mode=*/5);
  // t2 = g1 @ Wg2   (store transposed bf16)
  k_gemm<<<gemm_blocks(N_SEQ, OG_DIM), 256, 0, stream>>>(
      g1b, wg2t, N_SEQ, OG_DIM, HG_DIM, nullptr, nullptr, t2t, /*mode=*/8);
  // g2 = adj @ t2   (store f32)
  k_gemm<<<gemm_blocks(N_SEQ, OG_DIM), 256, 0, stream>>>(
      adjb, t2t, N_SEQ, OG_DIM, N_SEQ, g2, nullptr, nullptr, /*mode=*/2);

  // --- final projection ---
  k_fc<<<(N_SEQ + 255) / 256, 256, 0, stream>>>(g2, Wfc, bfc, (float*)d_out);
}
